// EnhancedGATNet_41506563949071
// MI455X (gfx1250) — matrix operand order repaired and checked by reading, not compiled, not run on hardware
//
#include <hip/hip_runtime.h>
#include <math.h>

// ---------------------------------------------------------------------------
// Problem constants (from reference)
// ---------------------------------------------------------------------------
#define NNODE 16000
#define NEDGE 64000
#define NB    256
#define NH    8
#define NC    128
#define HD    1024     // NH*NC
#define FXD   72
#define OUTD  512
#define ESMD  2560
#define ECFPD 2048
#define EPSV  1e-5f

// ---------------------------------------------------------------------------
// WMMA types (CDNA5 / gfx1250, wave32)
// ---------------------------------------------------------------------------
typedef __attribute__((ext_vector_type(16))) __bf16 v16bf;
typedef __attribute__((ext_vector_type(8)))  float  v8f;

union BFrag { unsigned int u[8]; v16bf bf; };

__device__ __forceinline__ unsigned short f32_to_bf16_rne(float f) {
  unsigned int x = __float_as_uint(f);
  unsigned int r = x + 0x7FFFu + ((x >> 16) & 1u);
  return (unsigned short)(r >> 16);
}

__device__ __forceinline__ unsigned int pack_bf16(float a, float b) {
  return (unsigned int)f32_to_bf16_rne(a) |
         ((unsigned int)f32_to_bf16_rne(b) << 16);
}

__device__ __forceinline__ float act_apply(float x, int act) {
  switch (act) {
    case 1: return 0.5f * x * (1.0f + erff(x * 0.7071067811865475f)); // gelu exact
    case 2: return x > 0.f ? x : expm1f(x);                           // elu
    case 3: return x / (1.0f + expf(-x));                             // silu
    case 4: return 1.0f / (1.0f + expf(-x));                          // sigmoid
    default: return x;
  }
}

__device__ __forceinline__ void atomicMaxF(float* addr, float val) {
  unsigned int* p = reinterpret_cast<unsigned int*>(addr);
  unsigned int old = *p;
  while (__uint_as_float(old) < val) {
    unsigned int assumed = old;
    old = atomicCAS(p, assumed, __float_as_uint(val));
    if (old == assumed) break;
  }
}

// ---------------------------------------------------------------------------
// WMMA GEMM:  C[M,N] = act( A[M,K] @ W[N,K]^T + bias + res )
// fp32 in/out, bf16 compute (v_wmma_f32_16x16x32_bf16), fp32 accumulate.
// Block tile 128x128x64, 256 threads = 8 waves; wave tile 32x64 (2x4 frags).
// Software pipelined: float4 global prefetch (LOADcnt) overlaps WMMA on the
// current LDS tile (DScnt); fragments read as ds_load_b128 (16B-aligned).
// ---------------------------------------------------------------------------
#define BM 128
#define BN 128
#define BK 64
#define LDT 72   // ushort stride: 144 B rows -> every fragment chunk 16B-aligned

__global__ __launch_bounds__(256)
void k_gemm(const float* __restrict__ A, const float* __restrict__ W,
            const float* __restrict__ bias, const float* __restrict__ res,
            float* __restrict__ C, int M, int N, int K, int act)
{
  __shared__ unsigned short sA[BM * LDT];   // [m][k]
  __shared__ unsigned short sB[BN * LDT];   // [n][k]  (B^T staged as W rows)

  const int tid = threadIdx.x;
  const int lane = tid & 31;
  const int wave = tid >> 5;
  const int wm   = wave >> 1;   // 0..3  -> 32-row slab
  const int wn   = wave & 1;    // 0..1  -> 64-col slab
  const int m0   = blockIdx.y * BM;
  const int n0   = blockIdx.x * BN;
  const int hl   = lane >> 4;
  const int l16  = lane & 15;

  v8f acc[2][4];
  #pragma unroll
  for (int i = 0; i < 2; ++i)
    #pragma unroll
    for (int j = 0; j < 4; ++j) {
      v8f z = {0.f,0.f,0.f,0.f,0.f,0.f,0.f,0.f};
      acc[i][j] = z;
    }

  // register staging: 128x64 tile = 2048 float4, 8 per thread (row = la>>4)
  float4 ra[8], rb[8];

  auto load_tiles = [&](int k0) {
    #pragma unroll
    for (int i = 0; i < 8; ++i) {
      int la = tid + 256 * i;
      int row = la >> 4, c4 = la & 15;
      int gk = k0 + c4 * 4;
      int gm = m0 + row;
      float4 v = {0.f, 0.f, 0.f, 0.f};
      if (gm < M && gk < K) {
        const float* p = A + (size_t)gm * K + gk;
        if (gk + 3 < K) v = *(const float4*)p;
        else {
          v.x = p[0];
          if (gk + 1 < K) v.y = p[1];
          if (gk + 2 < K) v.z = p[2];
        }
      }
      ra[i] = v;
      int gn = n0 + row;
      float4 w = {0.f, 0.f, 0.f, 0.f};
      if (gn < N && gk < K) {
        const float* p = W + (size_t)gn * K + gk;
        if (gk + 3 < K) w = *(const float4*)p;
        else {
          w.x = p[0];
          if (gk + 1 < K) w.y = p[1];
          if (gk + 2 < K) w.z = p[2];
        }
      }
      rb[i] = w;
    }
  };
  auto store_tiles = [&]() {
    #pragma unroll
    for (int i = 0; i < 8; ++i) {
      int la = tid + 256 * i;
      int row = la >> 4, c4 = la & 15;
      uint2 pa; pa.x = pack_bf16(ra[i].x, ra[i].y); pa.y = pack_bf16(ra[i].z, ra[i].w);
      *(uint2*)&sA[row * LDT + c4 * 4] = pa;
      uint2 pb; pb.x = pack_bf16(rb[i].x, rb[i].y); pb.y = pack_bf16(rb[i].z, rb[i].w);
      *(uint2*)&sB[row * LDT + c4 * 4] = pb;
    }
  };

  load_tiles(0);
  for (int k0 = 0;;) {
    __syncthreads();              // all waves done reading previous LDS tile
    store_tiles();
    __syncthreads();              // tile visible
    int kn = k0 + BK;
    bool more = kn < K;
    if (more) load_tiles(kn);     // global_load_b128 in flight during WMMAs

    #pragma unroll
    for (int kk = 0; kk < BK; kk += 32) {
      // A frag: 16-bit A 16x32 layout -> two contiguous 16B runs
      BFrag afr[2];
      #pragma unroll
      for (int mf = 0; mf < 2; ++mf) {
        int m = wm * 32 + mf * 16 + l16;
        const unsigned short* base = &sA[m * LDT + kk + hl * 8];
        uint4* au = reinterpret_cast<uint4*>(afr[mf].u);
        au[0] = *(const uint4*)(base);
        au[1] = *(const uint4*)(base + 16);
      }
      // B frag: 16-bit B 32x16 layout -> two contiguous 16B runs (same k order as A)
      #pragma unroll
      for (int nf = 0; nf < 4; ++nf) {
        int n = wn * 64 + nf * 16 + l16;
        const unsigned short* base = &sB[n * LDT + kk + hl * 8];
        BFrag bfr;
        uint4* bu = reinterpret_cast<uint4*>(bfr.u);
        bu[0] = *(const uint4*)(base);
        bu[1] = *(const uint4*)(base + 16);
        #pragma unroll
        for (int mf = 0; mf < 2; ++mf)
          acc[mf][nf] = __builtin_amdgcn_wmma_f32_16x16x32_bf16(
              false, afr[mf].bf, false, bfr.bf, (short)0, acc[mf][nf],
              false, false);
      }
    }
    if (!more) break;
    k0 = kn;
  }

  // epilogue: C/D layout m = v + 8*half
  for (int nf = 0; nf < 4; ++nf) {
    int n = n0 + wn * 64 + nf * 16 + l16;
    if (n >= N) continue;
    float bv = bias ? bias[n] : 0.f;
    #pragma unroll
    for (int mf = 0; mf < 2; ++mf) {
      #pragma unroll
      for (int vv = 0; vv < 8; ++vv) {
        int m = m0 + wm * 32 + mf * 16 + hl * 8 + vv;
        if (m >= M) continue;
        float val = acc[mf][nf][vv] + bv;
        if (res) val += res[(size_t)m * N + n];
        C[(size_t)m * N + n] = act_apply(val, act);
      }
    }
  }
}

// ---------------------------------------------------------------------------
// Utility / elementwise kernels
// ---------------------------------------------------------------------------
__global__ void k_fill(float* p, float v, long n) {
  long i = (long)blockIdx.x * blockDim.x + threadIdx.x;
  if (i < n) p[i] = v;
}

__global__ void k_elu(float* p, long n) {
  long i = (long)blockIdx.x * blockDim.x + threadIdx.x;
  if (i < n) { float x = p[i]; p[i] = x > 0.f ? x : expm1f(x); }
}

__global__ void k_mul(float* a, const float* b, long n) {
  long i = (long)blockIdx.x * blockDim.x + threadIdx.x;
  if (i < n) a[i] *= b[i];
}

// ---------------------------------------------------------------------------
// TransformerConv edge attention (H=8, C=128, HD=1024)
// ---------------------------------------------------------------------------
__global__ void k_edge_score(const float* __restrict__ q, const float* __restrict__ k,
                             const float* __restrict__ eW, const float* __restrict__ ea,
                             const int* __restrict__ src, const int* __restrict__ dst,
                             float* __restrict__ score, float* __restrict__ smax)
{
  int gid  = blockIdx.x * blockDim.x + threadIdx.x;
  int wid  = gid >> 5, lane = gid & 31;
  if (wid >= NEDGE * NH) return;
  int j = wid >> 3, h = wid & 7;
  int s = src[j], d = dst[j];
  float a0 = ea[2 * j], a1 = ea[2 * j + 1];
  float sum = 0.f;
  #pragma unroll
  for (int i = 0; i < 4; ++i) {
    int dd = h * NC + lane + 32 * i;
    float e = a0 * eW[2 * dd] + a1 * eW[2 * dd + 1];
    sum += q[(size_t)d * HD + dd] * (k[(size_t)s * HD + dd] + e);
  }
  for (int o = 16; o > 0; o >>= 1) sum += __shfl_down(sum, o, 32);
  if (lane == 0) {
    float sc = sum * 0.08838834764831845f;  // 1/sqrt(128)
    score[j * NH + h] = sc;
    atomicMaxF(&smax[d * NH + h], sc);
  }
}

__global__ void k_edge_expdenom(const float* __restrict__ score, const float* __restrict__ smax,
                                const int* __restrict__ dst, float* __restrict__ ex,
                                float* __restrict__ den)
{
  int i = blockIdx.x * blockDim.x + threadIdx.x;
  if (i >= NEDGE * NH) return;
  int j = i >> 3, h = i & 7;
  int d = dst[j];
  float e = expf(score[i] - smax[d * NH + h]);
  ex[i] = e;
  atomicAdd(&den[d * NH + h], e);
}

__global__ void k_edge_scatter(const float* __restrict__ v, const float* __restrict__ eW,
                               const float* __restrict__ ea, const int* __restrict__ src,
                               const int* __restrict__ dst, const float* __restrict__ ex,
                               const float* __restrict__ den, float* __restrict__ att)
{
  int gid  = blockIdx.x * blockDim.x + threadIdx.x;
  int wid  = gid >> 5, lane = gid & 31;
  if (wid >= NEDGE * NH) return;
  int j = wid >> 3, h = wid & 7;
  int s = src[j], d = dst[j];
  float a0 = ea[2 * j], a1 = ea[2 * j + 1];
  float alpha = ex[j * NH + h] / (den[d * NH + h] + 1e-16f);
  #pragma unroll
  for (int i = 0; i < 4; ++i) {
    int dd = h * NC + lane + 32 * i;
    float e = a0 * eW[2 * dd] + a1 * eW[2 * dd + 1];
    atomicAdd(&att[(size_t)d * HD + dd], alpha * (v[(size_t)s * HD + dd] + e));
  }
}

__global__ void k_head_mean(const float* __restrict__ att, float* __restrict__ out) {
  long i = (long)blockIdx.x * blockDim.x + threadIdx.x;
  if (i >= (long)NNODE * NC) return;
  long n = i / NC; int c = (int)(i % NC);
  float s = 0.f;
  #pragma unroll
  for (int h = 0; h < NH; ++h) s += att[n * HD + h * NC + c];
  out[i] = s * (1.0f / NH);
}

// beta gating: beta = sigmoid([out, xr, out-xr] @ bw^T); h = beta*xr + (1-beta)*out
__global__ void k_beta(const float* __restrict__ out, const float* __restrict__ xr,
                       const float* __restrict__ bw, float* __restrict__ dstb, int d)
{
  int gid = blockIdx.x * blockDim.x + threadIdx.x;
  int n = gid >> 5, lane = gid & 31;
  if (n >= NNODE) return;
  float s = 0.f;
  for (int j = lane; j < d; j += 32) {
    float o = out[(size_t)n * d + j], x = xr[(size_t)n * d + j];
    s += o * bw[j] + x * bw[d + j] + (o - x) * bw[2 * d + j];
  }
  for (int o = 16; o > 0; o >>= 1) s += __shfl_down(s, o, 32);
  s = __shfl(s, 0, 32);
  float beta = 1.0f / (1.0f + expf(-s));
  for (int j = lane; j < d; j += 32) {
    float o = out[(size_t)n * d + j], x = xr[(size_t)n * d + j];
    dstb[(size_t)n * d + j] = beta * x + (1.0f - beta) * o;
  }
}

// ---------------------------------------------------------------------------
// graph LayerNorm (PyG mode='graph')
// ---------------------------------------------------------------------------
__global__ void k_gln_accum(const float* __restrict__ x, const int* __restrict__ batch,
                            float* __restrict__ gs, float* __restrict__ gss,
                            float* __restrict__ gc, int d)
{
  int gid = blockIdx.x * blockDim.x + threadIdx.x;
  int n = gid >> 5, lane = gid & 31;
  if (n >= NNODE) return;
  float s = 0.f, ss = 0.f;
  for (int j = lane; j < d; j += 32) {
    float v = x[(size_t)n * d + j];
    s += v; ss += v * v;
  }
  for (int o = 16; o > 0; o >>= 1) { s += __shfl_down(s, o, 32); ss += __shfl_down(ss, o, 32); }
  if (lane == 0) {
    int g = batch[n];
    atomicAdd(&gs[g], s);
    atomicAdd(&gss[g], ss);
    atomicAdd(&gc[g], 1.0f);
  }
}

__global__ void k_gln_apply(float* __restrict__ x, const int* __restrict__ batch,
                            const float* __restrict__ gs, const float* __restrict__ gss,
                            const float* __restrict__ gc, const float* __restrict__ w,
                            const float* __restrict__ b, int d)
{
  long i = (long)blockIdx.x * blockDim.x + threadIdx.x;
  if (i >= (long)NNODE * d) return;
  long n = i / d; int j = (int)(i % d);
  int g = batch[n];
  float cnt = gc[g] < 1.f ? 1.f : gc[g];
  float norm = cnt * (float)d;
  float mean = gs[g] / norm;
  float var  = gss[g] / norm - mean * mean;
  if (var < 0.f) var = 0.f;
  x[i] = (x[i] - mean) * rsqrtf(var + EPSV) * w[j] + b[j];
}

// ---------------------------------------------------------------------------
// Row LayerNorm (+ optional activation)
// ---------------------------------------------------------------------------
__global__ void k_rowln(const float* __restrict__ x, const float* __restrict__ w,
                        const float* __restrict__ b, float* __restrict__ out,
                        int rows, int d, int act)
{
  int gid = blockIdx.x * blockDim.x + threadIdx.x;
  int r = gid >> 5, lane = gid & 31;
  if (r >= rows) return;
  float s = 0.f, ss = 0.f;
  for (int j = lane; j < d; j += 32) {
    float v = x[(size_t)r * d + j];
    s += v; ss += v * v;
  }
  for (int o = 16; o > 0; o >>= 1) { s += __shfl_down(s, o, 32); ss += __shfl_down(ss, o, 32); }
  s = __shfl(s, 0, 32); ss = __shfl(ss, 0, 32);
  float mean = s / d;
  float var = ss / d - mean * mean;
  if (var < 0.f) var = 0.f;
  float inv = rsqrtf(var + EPSV);
  for (int j = lane; j < d; j += 32) {
    float v = (x[(size_t)r * d + j] - mean) * inv * w[j] + b[j];
    out[(size_t)r * d + j] = act_apply(v, act);
  }
}

// ---------------------------------------------------------------------------
// Set2Set (B=256, C=128)
// ---------------------------------------------------------------------------
__global__ void k_lstm_gates(const float* __restrict__ qst, const float* __restrict__ hst,
                             const float* __restrict__ wih, const float* __restrict__ whh,
                             const float* __restrict__ bih, const float* __restrict__ bhh,
                             float* __restrict__ g)
{
  int i = blockIdx.x * blockDim.x + threadIdx.x;
  if (i >= NB * 512) return;
  int bb = i >> 9, o = i & 511;
  float s = bih[o] + bhh[o];
  const float* qr = qst + bb * 256;
  const float* wr = wih + o * 256;
  for (int j = 0; j < 256; ++j) s += qr[j] * wr[j];
  const float* hr = hst + bb * NC;
  const float* vr = whh + o * NC;
  for (int j = 0; j < NC; ++j) s += hr[j] * vr[j];
  g[i] = s;
}

__global__ void k_lstm_update(const float* __restrict__ g, float* __restrict__ cst,
                              float* __restrict__ hst)
{
  int i = blockIdx.x * blockDim.x + threadIdx.x;
  if (i >= NB * NC) return;
  int bb = i >> 7, j = i & 127;
  const float* gr = g + bb * 512;
  float gi = 1.f / (1.f + expf(-gr[j]));
  float gf = 1.f / (1.f + expf(-gr[128 + j]));
  float gg = tanhf(gr[256 + j]);
  float go = 1.f / (1.f + expf(-gr[384 + j]));
  float c = gf * cst[i] + gi * gg;
  cst[i] = c;
  hst[i] = go * tanhf(c);
}

__global__ void k_s2s_energy(const float* __restrict__ x, const int* __restrict__ batch,
                             const float* __restrict__ hst, float* __restrict__ en,
                             float* __restrict__ emax)
{
  int gid = blockIdx.x * blockDim.x + threadIdx.x;
  int n = gid >> 5, lane = gid & 31;
  if (n >= NNODE) return;
  int g = batch[n];
  float s = 0.f;
  #pragma unroll
  for (int i = 0; i < 4; ++i) {
    int c = lane + 32 * i;
    s += x[(size_t)n * NC + c] * hst[g * NC + c];
  }
  for (int o = 16; o > 0; o >>= 1) s += __shfl_down(s, o, 32);
  if (lane == 0) { en[n] = s; atomicMaxF(&emax[g], s); }
}

__global__ void k_s2s_fixmax(float* emax) {
  int g = blockIdx.x * blockDim.x + threadIdx.x;
  if (g < NB && emax[g] == -__builtin_inff()) emax[g] = 0.f;
}

__global__ void k_s2s_expsum(float* __restrict__ en, const float* __restrict__ emax,
                             const int* __restrict__ batch, float* __restrict__ esum)
{
  int n = blockIdx.x * blockDim.x + threadIdx.x;
  if (n >= NNODE) return;
  float ae = expf(en[n] - emax[batch[n]]);
  en[n] = ae;
  atomicAdd(&esum[batch[n]], ae);
}

__global__ void k_s2s_r(const float* __restrict__ x, const float* __restrict__ en,
                        const float* __restrict__ esum, const int* __restrict__ batch,
                        float* __restrict__ r)
{
  int gid = blockIdx.x * blockDim.x + threadIdx.x;
  int n = gid >> 5, lane = gid & 31;
  if (n >= NNODE) return;
  int g = batch[n];
  float a = en[n] / (esum[g] + 1e-16f);
  #pragma unroll
  for (int i = 0; i < 4; ++i) {
    int c = lane + 32 * i;
    atomicAdd(&r[g * NC + c], a * x[(size_t)n * NC + c]);
  }
}

__global__ void k_s2s_qstar(const float* __restrict__ hst, const float* __restrict__ r,
                            float* __restrict__ qst)
{
  int i = blockIdx.x * blockDim.x + threadIdx.x;
  if (i >= NB * 256) return;
  int bb = i >> 8, j = i & 255;
  qst[i] = (j < NC) ? hst[bb * NC + j] : r[bb * NC + (j - NC)];
}

// ---------------------------------------------------------------------------
// Fusion + final dot
// ---------------------------------------------------------------------------
__global__ void k_fusion(const float* __restrict__ xgnn, const float* __restrict__ xecfp,
                         const float* __restrict__ xprot, const float* __restrict__ gate,
                         const float* __restrict__ fw, float* __restrict__ xc)
{
  int i = blockIdx.x * blockDim.x + threadIdx.x;
  if (i >= NB * OUTD) return;
  int bb = i / OUTD, j = i % OUTD;
  float f0 = fw[0], f1 = fw[1], f2 = fw[2];
  float mx = fmaxf(f0, fmaxf(f1, f2));
  float e0 = expf(f0 - mx), e1 = expf(f1 - mx), e2 = expf(f2 - mx);
  float inv = 1.f / (e0 + e1 + e2);
  float w0 = e0 * inv, w1 = e1 * inv, w2 = e2 * inv;
  float g  = gate[i];
  float vg = xgnn[i], ve = xecfp[i], vp = xprot[i];
  float xg = vg * g, xe = ve * g;
  float fused = w0 * xg + w1 * xe + w2 * vp;
  float attn  = (vg + ve + vp) * (1.0f / 3.0f);
  float* row = xc + (size_t)bb * (5 * OUTD);
  row[j] = xg;
  row[OUTD + j] = xe;
  row[2 * OUTD + j] = vp;
  row[3 * OUTD + j] = fused;
  row[4 * OUTD + j] = attn;
}

__global__ void k_dot_out(const float* __restrict__ y, const float* __restrict__ w,
                          const float* __restrict__ b, float* __restrict__ out)
{
  int gid = blockIdx.x * blockDim.x + threadIdx.x;
  int r = gid >> 5, lane = gid & 31;
  if (r >= NB) return;
  float s = 0.f;
  for (int j = lane; j < OUTD; j += 32) s += y[(size_t)r * OUTD + j] * w[j];
  for (int o = 16; o > 0; o >>= 1) s += __shfl_down(s, o, 32);
  if (lane == 0) out[r] = s + b[0];
}

// ---------------------------------------------------------------------------
// Parameter leaf indices (JAX pytree order: dict keys sorted at every level)
// ---------------------------------------------------------------------------
enum {
  CLF_L1_W = 0, CLF_L1_B, CLF_L2_W, CLF_L2_B, CLF_LN_B, CLF_LN_W,
  CLF_RES_L1_W, CLF_RES_L1_B, CLF_RES_L2_W, CLF_RES_L2_B, CLF_RES_LN_B, CLF_RES_LN_W,
  CLF_SW_W1_W, CLF_SW_W1_B, CLF_SW_W2_W, CLF_SW_W2_B, CLF_SW_W3_W, CLF_SW_W3_B,
  C1_BETA_W, C1_E_W, C1_K_W, C1_K_B, C1_Q_W, C1_Q_B, C1_SKIP_W, C1_SKIP_B, C1_V_W, C1_V_B,
  C2_BETA_W, C2_E_W, C2_K_W, C2_K_B, C2_Q_W, C2_Q_B, C2_SKIP_W, C2_SKIP_B, C2_V_W, C2_V_B,
  ECFP_LN_B, ECFP_LN_W, ECFP_OUT_W, ECFP_OUT_B, ECFP_RED_W, ECFP_RED_B,
  FC_G1_W, FC_G1_B, FC_G2_W, FC_G2_B, FUSION_W, GATE_W, GATE_B,
  LN1_B, LN1_W, LN2_B, LN2_W,
  LSTM_W_HH, LSTM_W_IH, LSTM_B_HH, LSTM_B_IH,
  PROT_IN_W1_W, PROT_IN_W1_B, PROT_IN_W2_W, PROT_IN_W2_B, PROT_IN_W3_W, PROT_IN_W3_B,
  PROT_LN_B, PROT_LN_W, PROT_OUT_W, PROT_OUT_B,
  PROT_R1_L1_W, PROT_R1_L1_B, PROT_R1_L2_W, PROT_R1_L2_B, PROT_R1_LN_B, PROT_R1_LN_W,
  PROT_R2_L1_W, PROT_R2_L1_B, PROT_R2_L2_W, PROT_R2_L2_B, PROT_R2_LN_B, PROT_R2_LN_W,
  PARAM_COUNT
};

static inline int cdiv(long a, long b) { return (int)((a + b - 1) / b); }

extern "C" void kernel_launch(void* const* d_in, const int* in_sizes, int n_in,
                              void* d_out, int out_size, void* d_ws, size_t ws_size,
                              hipStream_t stream)
{
  (void)in_sizes; (void)n_in; (void)out_size; (void)ws_size;

  const float* x     = (const float*)d_in[0];
  const int*   eidx  = (const int*)d_in[1];
  const float* eattr = (const float*)d_in[2];
  const int*   batch = (const int*)d_in[3];
  const float* ecfp  = (const float*)d_in[4];
  const float* protF = (const float*)d_in[5];
  const float* const* PP = reinterpret_cast<const float* const*>(d_in + 6);

  const int* src = eidx;           // edge_index[0]
  const int* dst = eidx + NEDGE;   // edge_index[1]

  float* ws = (float*)d_ws;
  size_t off = 0;
  auto alloc = [&](size_t n) { float* p = ws + off; off += n; return p; };

  float* Q    = alloc((size_t)NNODE * HD);
  float* Kb   = alloc((size_t)NNODE * HD);
  float* Vb   = alloc((size_t)NNODE * HD);
  float* SK   = alloc((size_t)NNODE * HD);   // conv1 skip (1024) / conv2 skip (128)
  float* ATT  = alloc((size_t)NNODE * HD);
  float* H1   = alloc((size_t)NNODE * HD);
  float* H2   = alloc((size_t)NNODE * NC);
  float* MEANB= alloc((size_t)NNODE * NC);
  float* SCORE= alloc((size_t)NEDGE * NH);
  float* EXB  = alloc((size_t)NEDGE * NH);
  float* SMAX = alloc((size_t)NNODE * NH);
  float* DEN  = alloc((size_t)NNODE * NH);
  float* GS   = alloc(NB);
  float* GSS  = alloc(NB);
  float* GC   = alloc(NB);
  float* HST  = alloc((size_t)NB * NC);
  float* CST  = alloc((size_t)NB * NC);
  float* QST  = alloc((size_t)NB * 256);
  float* G4   = alloc((size_t)NB * 512);
  float* EN   = alloc(NNODE);
  float* EMX  = alloc(NB);
  float* ESUM = alloc(NB);
  float* RB   = alloc((size_t)NB * NC);
  float* T1   = alloc((size_t)NB * 1024);
  float* T2   = alloc((size_t)NB * 1024);
  float* PR   = alloc((size_t)NB * 1024);
  float* HB   = alloc((size_t)NB * 1024);
  float* XGNN = alloc((size_t)NB * OUTD);
  float* E1   = alloc((size_t)NB * 1024);
  float* XECFP= alloc((size_t)NB * OUTD);
  float* XP0  = alloc((size_t)NB * OUTD);
  float* XPROT= alloc((size_t)NB * OUTD);
  float* GATE = alloc((size_t)NB * OUTD);
  float* XC   = alloc((size_t)NB * 5 * OUTD);
  float* Y    = alloc((size_t)NB * 1024);
  float* Y2   = alloc((size_t)NB * 1024);
  float* L1   = alloc((size_t)NB * OUTD);
  float* Y3   = alloc((size_t)NB * OUTD);

  const int T = 256;
  auto gemm = [&](const float* A, const float* W, const float* bias, const float* res,
                  float* C, int M, int N, int K, int act) {
    dim3 g(cdiv(N, BN), cdiv(M, BM));
    k_gemm<<<g, T, 0, stream>>>(A, W, bias, res, C, M, N, K, act);
  };
  auto fill = [&](float* p, float v, long n) {
    k_fill<<<cdiv(n, T), T, 0, stream>>>(p, v, n);
  };

  const float NEG_INF = -__builtin_inff();
  const int nodeWaves = cdiv((long)NNODE * 32, T);
  const int ehWaves   = cdiv((long)NEDGE * NH * 32, T);

  auto attention = [&](const float* qp, const float* kp, const float* vp,
                       const float* eW) {
    fill(SMAX, NEG_INF, (long)NNODE * NH);
    fill(DEN, 0.f, (long)NNODE * NH);
    fill(ATT, 0.f, (long)NNODE * HD);
    k_edge_score<<<ehWaves, T, 0, stream>>>(qp, kp, eW, eattr, src, dst, SCORE, SMAX);
    k_edge_expdenom<<<cdiv((long)NEDGE * NH, T), T, 0, stream>>>(SCORE, SMAX, dst, EXB, DEN);
    k_edge_scatter<<<ehWaves, T, 0, stream>>>(vp, eW, eattr, src, dst, EXB, DEN, ATT);
  };
  auto graph_ln = [&](float* hp, const float* w, const float* b, int d) {
    fill(GS, 0.f, NB); fill(GSS, 0.f, NB); fill(GC, 0.f, NB);
    k_gln_accum<<<nodeWaves, T, 0, stream>>>(hp, batch, GS, GSS, GC, d);
    k_gln_apply<<<cdiv((long)NNODE * d, T), T, 0, stream>>>(hp, batch, GS, GSS, GC, w, b, d);
  };

  // ---------------- conv1 (in=72, heads concat -> 1024) ----------------
  gemm(x, PP[C1_Q_W], PP[C1_Q_B], nullptr, Q, NNODE, HD, FXD, 0);
  gemm(x, PP[C1_K_W], PP[C1_K_B], nullptr, Kb, NNODE, HD, FXD, 0);
  gemm(x, PP[C1_V_W], PP[C1_V_B], nullptr, Vb, NNODE, HD, FXD, 0);
  gemm(x, PP[C1_SKIP_W], PP[C1_SKIP_B], nullptr, SK, NNODE, HD, FXD, 0);
  attention(Q, Kb, Vb, PP[C1_E_W]);
  k_beta<<<nodeWaves, T, 0, stream>>>(ATT, SK, PP[C1_BETA_W], H1, HD);
  k_elu<<<cdiv((long)NNODE * HD, T), T, 0, stream>>>(H1, (long)NNODE * HD);
  graph_ln(H1, PP[LN1_W], PP[LN1_B], HD);

  // ---------------- conv2 (in=1024, head-mean -> 128) ----------------
  gemm(H1, PP[C2_Q_W], PP[C2_Q_B], nullptr, Q, NNODE, HD, HD, 0);
  gemm(H1, PP[C2_K_W], PP[C2_K_B], nullptr, Kb, NNODE, HD, HD, 0);
  gemm(H1, PP[C2_V_W], PP[C2_V_B], nullptr, Vb, NNODE, HD, HD, 0);
  gemm(H1, PP[C2_SKIP_W], PP[C2_SKIP_B], nullptr, SK, NNODE, NC, HD, 0);
  attention(Q, Kb, Vb, PP[C2_E_W]);
  k_head_mean<<<cdiv((long)NNODE * NC, T), T, 0, stream>>>(ATT, MEANB);
  k_beta<<<nodeWaves, T, 0, stream>>>(MEANB, SK, PP[C2_BETA_W], H2, NC);
  k_elu<<<cdiv((long)NNODE * NC, T), T, 0, stream>>>(H2, (long)NNODE * NC);
  graph_ln(H2, PP[LN2_W], PP[LN2_B], NC);

  // ---------------- set2set (3 steps) ----------------
  fill(HST, 0.f, (long)NB * NC);
  fill(CST, 0.f, (long)NB * NC);
  fill(QST, 0.f, (long)NB * 256);
  for (int step = 0; step < 3; ++step) {
    k_lstm_gates<<<cdiv((long)NB * 512, T), T, 0, stream>>>(
        QST, HST, PP[LSTM_W_IH], PP[LSTM_W_HH], PP[LSTM_B_IH], PP[LSTM_B_HH], G4);
    k_lstm_update<<<cdiv((long)NB * NC, T), T, 0, stream>>>(G4, CST, HST);
    fill(EMX, NEG_INF, NB); fill(ESUM, 0.f, NB); fill(RB, 0.f, (long)NB * NC);
    k_s2s_energy<<<nodeWaves, T, 0, stream>>>(H2, batch, HST, EN, EMX);
    k_s2s_fixmax<<<1, NB, 0, stream>>>(EMX);
    k_s2s_expsum<<<cdiv(NNODE, T), T, 0, stream>>>(EN, EMX, batch, ESUM);
    k_s2s_r<<<nodeWaves, T, 0, stream>>>(H2, EN, ESUM, batch, RB);
    k_s2s_qstar<<<cdiv((long)NB * 256, T), T, 0, stream>>>(HST, RB, QST);
  }

  // ---------------- GNN head ----------------
  gemm(QST, PP[FC_G1_W], PP[FC_G1_B], nullptr, T1, NB, 1024, 256, 2);   // elu
  gemm(T1, PP[FC_G2_W], PP[FC_G2_B], nullptr, XGNN, NB, OUTD, 1024, 0);

  // ---------------- ECFP branch ----------------
  gemm(ecfp, PP[ECFP_RED_W], PP[ECFP_RED_B], nullptr, E1, NB, 1024, ECFPD, 0);
  k_rowln<<<cdiv((long)NB * 32, T), T, 0, stream>>>(E1, PP[ECFP_LN_W], PP[ECFP_LN_B],
                                                    E1, NB, 1024, 1);   // gelu
  gemm(E1, PP[ECFP_OUT_W], PP[ECFP_OUT_B], nullptr, XECFP, NB, OUTD, 1024, 1); // gelu

  // ---------------- Protein branch ----------------
  gemm(protF, PP[PROT_IN_W1_W], PP[PROT_IN_W1_B], nullptr, T1, NB, 1024, ESMD, 3); // silu
  gemm(protF, PP[PROT_IN_W2_W], PP[PROT_IN_W2_B], nullptr, T2, NB, 1024, ESMD, 0);
  k_mul<<<cdiv((long)NB * 1024, T), T, 0, stream>>>(T1, T2, (long)NB * 1024);
  gemm(T1, PP[PROT_IN_W3_W], PP[PROT_IN_W3_B], nullptr, PR, NB, 1024, 1024, 0);
  // res1
  k_rowln<<<cdiv((long)NB * 32, T), T, 0, stream>>>(PR, PP[PROT_R1_LN_W], PP[PROT_R1_LN_B],
                                                    HB, NB, 1024, 0);
  gemm(HB, PP[PROT_R1_L1_W], PP[PROT_R1_L1_B], nullptr, T1, NB, 1024, 1024, 1); // gelu
  gemm(T1, PP[PROT_R1_L2_W], PP[PROT_R1_L2_B], PR, PR, NB, 1024, 1024, 0);      // +res
  // res2
  k_rowln<<<cdiv((long)NB * 32, T), T, 0, stream>>>(PR, PP[PROT_R2_LN_W], PP[PROT_R2_LN_B],
                                                    HB, NB, 1024, 0);
  gemm(HB, PP[PROT_R2_L1_W], PP[PROT_R2_L1_B], nullptr, T1, NB, 1024, 1024, 1);
  gemm(T1, PP[PROT_R2_L2_W], PP[PROT_R2_L2_B], PR, PR, NB, 1024, 1024, 0);
  gemm(PR, PP[PROT_OUT_W], PP[PROT_OUT_B], nullptr, XP0, NB, OUTD, 1024, 0);
  k_rowln<<<cdiv((long)NB * 32, T), T, 0, stream>>>(XP0, PP[PROT_LN_W], PP[PROT_LN_B],
                                                    XPROT, NB, OUTD, 0);

  // ---------------- Fusion ----------------
  gemm(XPROT, PP[GATE_W], PP[GATE_B], nullptr, GATE, NB, OUTD, OUTD, 4); // sigmoid
  k_fusion<<<cdiv((long)NB * OUTD, T), T, 0, stream>>>(XGNN, XECFP, XPROT, GATE,
                                                       PP[FUSION_W], XC);

  // ---------------- Classifier ----------------
  gemm(XC, PP[CLF_SW_W1_W], PP[CLF_SW_W1_B], nullptr, T1, NB, 1024, 5 * OUTD, 3); // silu
  gemm(XC, PP[CLF_SW_W2_W], PP[CLF_SW_W2_B], nullptr, T2, NB, 1024, 5 * OUTD, 0);
  k_mul<<<cdiv((long)NB * 1024, T), T, 0, stream>>>(T1, T2, (long)NB * 1024);
  gemm(T1, PP[CLF_SW_W3_W], PP[CLF_SW_W3_B], nullptr, Y, NB, 1024, 1024, 0);
  k_rowln<<<cdiv((long)NB * 32, T), T, 0, stream>>>(Y, PP[CLF_RES_LN_W], PP[CLF_RES_LN_B],
                                                    HB, NB, 1024, 0);
  gemm(HB, PP[CLF_RES_L1_W], PP[CLF_RES_L1_B], nullptr, T1, NB, 1024, 1024, 1);
  gemm(T1, PP[CLF_RES_L2_W], PP[CLF_RES_L2_B], Y, Y2, NB, 1024, 1024, 0);
  gemm(Y2, PP[CLF_L1_W], PP[CLF_L1_B], nullptr, L1, NB, OUTD, 1024, 0);
  k_rowln<<<cdiv((long)NB * 32, T), T, 0, stream>>>(L1, PP[CLF_LN_W], PP[CLF_LN_B],
                                                    Y3, NB, OUTD, 2);   // elu
  k_dot_out<<<cdiv((long)NB * 32, T), T, 0, stream>>>(Y3, PP[CLF_L2_W], PP[CLF_L2_B],
                                                      (float*)d_out);
}
